// MambaVisionCDDecoderBlock_42082089566727
// MI455X (gfx1250) — matrix operand order repaired and checked by reading, not compiled, not run on hardware
//
#include <hip/hip_runtime.h>

typedef __attribute__((ext_vector_type(16))) __bf16 bf16x16;
typedef __attribute__((ext_vector_type(8)))  __bf16 bf16x8;
typedef __attribute__((ext_vector_type(8)))  float  f32x8;

namespace {

constexpr int NB   = 8;         // batch
constexpr int DM   = 192;       // d_model / d_inner
constexpr int DX   = 96;        // x/z branch width
constexpr int NST  = 16;        // d_state
constexpr int DTR  = 12;        // dt_rank
constexpr int HH   = 56;
constexpr int WWI  = 56;
constexpr int LL   = HH * WWI;  // 3136
constexpr int OC   = 96;        // out channels
constexpr int OH   = 112;
constexpr int OWD  = 112;
constexpr int NT64 = LL / 64;   // 49 column groups (wave does 16x64)
constexpr int JP   = 48;        // padded x_dbl row count (44 -> 48)

__device__ __forceinline__ bf16x16 concat8(bf16x8 lo, bf16x8 hi) {
  return __builtin_shufflevector(lo, hi, 0, 1, 2, 3, 4, 5, 6, 7,
                                 8, 9, 10, 11, 12, 13, 14, 15);
}

// A: row-major [M x K] bf16 weights. Contiguous K layout: lane<16 -> kb=k0,
// lane>=16 -> kb=k0+16; elements 0..15 = K kb..kb+15.
__device__ __forceinline__ bf16x16 load_a_bf16(const __bf16* __restrict__ A, int lda,
                                               int row0, int k0, int lane) {
  const __bf16* p = A + (size_t)(row0 + (lane & 15)) * lda + k0 + (lane & 16);
  return concat8(*(const bf16x8*)(p), *(const bf16x8*)(p + 8));
}

// B operand from channel-last activations X[L][C]: lane column = col0+(lane&15),
// K-half = (lane&16); element e holds K = half+e -> 32 contiguous bytes.
__device__ __forceinline__ bf16x16 load_b_kmaj(const __bf16* __restrict__ X, int ldx,
                                               int col0, int k0, int lane) {
  const __bf16* p = X + (size_t)(col0 + (lane & 15)) * ldx + k0 + (lane & 16);
  return concat8(*(const bf16x8*)(p), *(const bf16x8*)(p + 8));
}

#define WMMA_BF16(acc, af, bfr)                                                      \
  acc = __builtin_amdgcn_wmma_f32_16x16x32_bf16(false, af, false, bfr, (short)0,     \
                                                acc, false, false)

// --------------------------- pre-pass converters -----------------------------
__global__ void k_cvt(const float* __restrict__ s, __bf16* __restrict__ d, int n) {
  const int i = blockIdx.x * 256 + threadIdx.x;
  if (i < n) d[i] = (__bf16)s[i];
}

// x [B][C][L] fp32 -> xbf [B][L][C] bf16 (channel-last for K-major B loads)
__global__ void k_cvtT_x(const float* __restrict__ x, __bf16* __restrict__ o) {
  const size_t i = (size_t)blockIdx.x * 256 + threadIdx.x;
  if (i >= (size_t)NB * LL * DM) return;
  const int c = (int)(i % DM);
  const int l = (int)((i / DM) % LL);
  const int b = (int)(i / ((size_t)DM * LL));
  o[i] = (__bf16)x[((size_t)b * DM + c) * LL + l];
}

// x_proj_w [44,96] -> bf16 [48,96], rows 44..47 zero
__global__ void k_pack_xproj(const float* __restrict__ W, __bf16* __restrict__ o) {
  const int i = blockIdx.x * 256 + threadIdx.x;
  if (i >= JP * DX) return;
  const int j = i / DX, d = i % DX;
  o[i] = (__bf16)(j < 44 ? W[j * DX + d] : 0.f);
}

// conv2d_w [co][ci][ky][kx] -> tap-major A [co][r*DM+ci], r = ky*3+kx
__global__ void k_pack_c2(const float* __restrict__ W, __bf16* __restrict__ o) {
  const int i = blockIdx.x * 256 + threadIdx.x;
  if (i >= DM * 9 * DM) return;
  const int co = i / (9 * DM);
  const int k  = i % (9 * DM);
  const int r  = k / DM;
  const int ci = k % DM;
  o[i] = (__bf16)W[((size_t)co * DM + ci) * 9 + r];
}

// convT weights per parity class s=(sy,sx), tap-major:
// o[s][co][jj*DM+ci] = convT_w[ci][co][3-ky][3-kx], ky=sy+2*(jj>>1), kx=sx+2*(jj&1)
__global__ void k_pack_convT(const float* __restrict__ Wt, __bf16* __restrict__ o) {
  const int i = blockIdx.x * 256 + threadIdx.x;
  if (i >= 4 * OC * 4 * DM) return;
  const int s  = i / (OC * 4 * DM);
  const int r  = i % (OC * 4 * DM);
  const int co = r / (4 * DM);
  const int k  = r % (4 * DM);
  const int jj = k / DM;
  const int ci = k % DM;
  const int sy = s >> 1, sx = s & 1;
  const int ky = sy + 2 * (jj >> 1);
  const int kx = sx + 2 * (jj & 1);
  o[i] = (__bf16)Wt[((size_t)ci * OC + co) * 16 + (3 - ky) * 4 + (3 - kx)];
}

// ---------- in_proj: xz[b,l,n] = sum_d W[n,d] * xbf[b,l,d] (channel-last) -----
__global__ void k_inproj(const __bf16* __restrict__ xb, const __bf16* __restrict__ W,
                         float* __restrict__ xz) {
  const int lane = threadIdx.x & 31;
  const int tile = blockIdx.x * 4 + (threadIdx.x >> 5);
  const int TOT = NB * (DM / 16) * NT64;
  if (tile >= TOT) return;
  const int b  = tile / ((DM / 16) * NT64);
  const int r  = tile % ((DM / 16) * NT64);
  const int m0 = (r / NT64) * 16;
  const int n0 = (r % NT64) * 64;
  const __bf16* Bb = xb + (size_t)b * LL * DM;
  f32x8 acc[4] = {};
#pragma unroll
  for (int k0 = 0; k0 < DM; k0 += 32) {
    const bf16x16 af = load_a_bf16(W, DM, m0, k0, lane);
#pragma unroll
    for (int nn = 0; nn < 4; ++nn) {
      const bf16x16 bf = load_b_kmaj(Bb, DM, n0 + 16 * nn, k0, lane);
      WMMA_BF16(acc[nn], af, bf);
    }
  }
  const int rb = m0 + ((lane & 16) ? 8 : 0);
#pragma unroll
  for (int nn = 0; nn < 4; ++nn) {
    const int col = n0 + 16 * nn + (lane & 15);
    float* ob = xz + ((size_t)b * LL + col) * DM + rb;
#pragma unroll
    for (int rr = 0; rr < 8; ++rr) ob[rr] = acc[nn][rr];
  }
}

// ------------- depthwise conv1d (k=3, same) + SiLU, split xh / z -------------
// xh -> fp32 [B][L][DX] (scan) + bf16 [B][L][DX] (x_proj); z -> yz slot c>=96.
__global__ void k_dwconv(const float* __restrict__ xz,
                         const float* __restrict__ wx, const float* __restrict__ bx,
                         const float* __restrict__ wz, const float* __restrict__ bz,
                         float* __restrict__ xhf, __bf16* __restrict__ xhb,
                         __bf16* __restrict__ yz) {
  const size_t i = (size_t)blockIdx.x * 256 + threadIdx.x;
  if (i >= (size_t)NB * LL * DM) return;
  const int c = (int)(i % DM);
  const int l = (int)((i / DM) % LL);
  const int b = (int)(i / ((size_t)DM * LL));
  const float v0 = xz[i];
  const float vm = (l > 0)      ? xz[i - DM] : 0.f;
  const float vp = (l < LL - 1) ? xz[i + DM] : 0.f;
  if (c < DX) {
    const float v = vm * wx[c * 3] + v0 * wx[c * 3 + 1] + vp * wx[c * 3 + 2] + bx[c];
    const float s = v / (1.f + __expf(-v));
    const size_t o = ((size_t)b * LL + l) * DX + c;
    xhf[o] = s;
    xhb[o] = (__bf16)s;
  } else {
    const float v = vm * wz[(c - DX) * 3] + v0 * wz[(c - DX) * 3 + 1] +
                    vp * wz[(c - DX) * 3 + 2] + bz[c - DX];
    const float s = v / (1.f + __expf(-v));
    yz[i] = (__bf16)s;                      // z occupies channels 96..191
  }
}

// ------ x_proj: xdbl[b,l,j] = sum_d x_proj_w[j,d]*xh[b,l,d], 48 padded rows ---
__global__ void k_xproj(const __bf16* __restrict__ W, const __bf16* __restrict__ xhb,
                        float* __restrict__ xdbl) {
  const int lane = threadIdx.x & 31;
  const int tile = blockIdx.x * 4 + (threadIdx.x >> 5);
  const int MT = JP / 16;             // 3
  const int TOT = NB * MT * NT64;
  if (tile >= TOT) return;
  const int b  = tile / (MT * NT64);
  const int r  = tile % (MT * NT64);
  const int m0 = (r / NT64) * 16;
  const int n0 = (r % NT64) * 64;
  const __bf16* Bb = xhb + (size_t)b * LL * DX;
  f32x8 acc[4] = {};
#pragma unroll
  for (int k0 = 0; k0 < DX; k0 += 32) {
    const bf16x16 af = load_a_bf16(W, DX, m0, k0, lane);
#pragma unroll
    for (int nn = 0; nn < 4; ++nn) {
      const bf16x16 bf = load_b_kmaj(Bb, DX, n0 + 16 * nn, k0, lane);
      WMMA_BF16(acc[nn], af, bf);
    }
  }
  const int rb = m0 + ((lane & 16) ? 8 : 0);
#pragma unroll
  for (int nn = 0; nn < 4; ++nn) {
    const int col = n0 + 16 * nn + (lane & 15);
    float* ob = xdbl + ((size_t)b * LL + col) * JP + rb;
#pragma unroll
    for (int rr = 0; rr < 8; ++rr) ob[rr] = acc[nn][rr];
  }
}

// --------------------- dt = softplus(dt_r @ Wd^T + bd) -----------------------
__global__ void k_dt(const float* __restrict__ xdbl, const float* __restrict__ Wd,
                     const float* __restrict__ bd, float* __restrict__ dtc) {
  const size_t i = (size_t)blockIdx.x * 256 + threadIdx.x;
  if (i >= (size_t)NB * LL * DX) return;
  const int d = (int)(i % DX);
  const int l = (int)((i / DX) % LL);
  const int b = (int)(i / ((size_t)DX * LL));
  const float* row = xdbl + ((size_t)b * LL + l) * JP;
  float acc = bd[d];
#pragma unroll
  for (int r = 0; r < DTR; ++r) acc += row[r] * Wd[d * DTR + r];
  dtc[i] = (acc > 20.f) ? acc : log1pf(__expf(acc));
}

// -------- selective scan: one (b,d) channel per thread, 16 states in regs ----
// Lanes d=0..95 access adjacent channels at each l (coalesced, channel-last).
__global__ void k_scan(const float* __restrict__ dtc, const float* __restrict__ xhf,
                       const float* __restrict__ xdbl, const float* __restrict__ A_log,
                       const float* __restrict__ ssm_D, __bf16* __restrict__ yz) {
  const int b = blockIdx.x;
  const int d = threadIdx.x;
  if (d >= DX) return;
  float Ar[NST];
#pragma unroll
  for (int n = 0; n < NST; ++n) Ar[n] = -__expf(A_log[d * NST + n]);
  float h[NST];
#pragma unroll
  for (int n = 0; n < NST; ++n) h[n] = 0.f;
  const float Dd = ssm_D[d];
  const float* dtp = dtc + (size_t)b * LL * DX + d;
  const float* xp  = xhf + (size_t)b * LL * DX + d;
  const float* xd  = xdbl + (size_t)b * LL * JP;
  __bf16* yp = yz + (size_t)b * LL * DM + d;      // y occupies channels 0..95
  for (int l = 0; l < LL; ++l) {
    const float dt = dtp[(size_t)l * DX];
    const float xv = xp[(size_t)l * DX];
    const float dtx = dt * xv;
    const float* Brow = xd + (size_t)l * JP + DTR;        // j 12..27
    const float* Crow = Brow + NST;                       // j 28..43
    float y = 0.f;
#pragma unroll
    for (int n = 0; n < NST; ++n) {
      const float hn = __expf(dt * Ar[n]) * h[n] + dtx * Brow[n];
      h[n] = hn;
      y += hn * Crow[n];
    }
    yp[(size_t)l * DM] = (__bf16)(y + xv * Dd);
  }
}

// ---- out_proj over yz (y|z channel-concat) + x_last residual -> img_bf ------
__global__ void k_outproj(const __bf16* __restrict__ W, const __bf16* __restrict__ yz,
                          const float* __restrict__ xlast, __bf16* __restrict__ img) {
  const int lane = threadIdx.x & 31;
  const int tile = blockIdx.x * 4 + (threadIdx.x >> 5);
  const int TOT = NB * (DM / 16) * NT64;
  if (tile >= TOT) return;
  const int b  = tile / ((DM / 16) * NT64);
  const int r  = tile % ((DM / 16) * NT64);
  const int m0 = (r / NT64) * 16;
  const int n0 = (r % NT64) * 64;
  const __bf16* Bb = yz + (size_t)b * LL * DM;
  f32x8 acc[4] = {};
#pragma unroll
  for (int k0 = 0; k0 < DM; k0 += 32) {
    const bf16x16 af = load_a_bf16(W, DM, m0, k0, lane);
#pragma unroll
    for (int nn = 0; nn < 4; ++nn) {
      const bf16x16 bf = load_b_kmaj(Bb, DM, n0 + 16 * nn, k0, lane);
      WMMA_BF16(acc[nn], af, bf);
    }
  }
  const int rb = m0 + ((lane & 16) ? 8 : 0);
#pragma unroll
  for (int nn = 0; nn < 4; ++nn) {
    const int col = n0 + 16 * nn + (lane & 15);
    __bf16* ob = img + ((size_t)b * LL + col) * DM + rb;
#pragma unroll
    for (int rr = 0; rr < 8; ++rr)
      ob[rr] = (__bf16)(acc[nn][rr] + xlast[((size_t)b * DM + rb + rr) * LL + col]);
  }
}

// ------- conv2d 3x3 pad1 + bias + ReLU, tap-major implicit GEMM --------------
// K = r*DM+ci: each 32-wide slab sits in one tap -> one predicate per slab.
__global__ void k_conv2d(const __bf16* __restrict__ W9, const float* __restrict__ bias,
                         const __bf16* __restrict__ img, __bf16* __restrict__ y2) {
  const int lane = threadIdx.x & 31;
  const int tile = blockIdx.x * 4 + (threadIdx.x >> 5);
  const int TOT = NB * (DM / 16) * NT64;
  if (tile >= TOT) return;
  const int b  = tile / ((DM / 16) * NT64);
  const int r  = tile % ((DM / 16) * NT64);
  const int m0 = (r / NT64) * 16;
  const int n0 = (r % NT64) * 64;
  const int kh = (lane & 16);
  int oy[4], ox[4];
#pragma unroll
  for (int nn = 0; nn < 4; ++nn) {
    const int col = n0 + 16 * nn + (lane & 15);
    oy[nn] = col / WWI;
    ox[nn] = col % WWI;
  }
  const __bf16* ib = img + (size_t)b * LL * DM;
  const __bf16* pa = W9 + (size_t)(m0 + (lane & 15)) * (9 * DM);
  f32x8 acc[4] = {};
#pragma unroll 1
  for (int r9 = 0; r9 < 9; ++r9) {
    if (r9 < 8) __builtin_prefetch(pa + (r9 + 1) * DM, 0, 0);
    const int dy = r9 / 3 - 1, dx = r9 % 3 - 1;
    const __bf16* pb[4];
    bool ok[4];
#pragma unroll
    for (int nn = 0; nn < 4; ++nn) {
      const int iy = oy[nn] + dy, ix = ox[nn] + dx;
      ok[nn] = ((unsigned)iy < (unsigned)HH) && ((unsigned)ix < (unsigned)WWI);
      pb[nn] = ib + ((size_t)(iy * WWI + ix)) * DM + kh;
    }
#pragma unroll
    for (int c0 = 0; c0 < DM; c0 += 32) {
      const bf16x16 af = load_a_bf16(W9, 9 * DM, m0, r9 * DM + c0, lane);
#pragma unroll
      for (int nn = 0; nn < 4; ++nn) {
        bf16x16 bv = {};
        if (ok[nn])
          bv = concat8(*(const bf16x8*)(pb[nn] + c0), *(const bf16x8*)(pb[nn] + c0 + 8));
        WMMA_BF16(acc[nn], af, bv);
      }
    }
  }
  const int rb = m0 + ((lane & 16) ? 8 : 0);
#pragma unroll
  for (int nn = 0; nn < 4; ++nn) {
    const int col = n0 + 16 * nn + (lane & 15);
    __bf16* ob = y2 + ((size_t)b * LL + col) * DM + rb;
#pragma unroll
    for (int rr = 0; rr < 8; ++rr) {
      const float v = acc[nn][rr] + bias[rb + rr];
      ob[rr] = (__bf16)(v > 0.f ? v : 0.f);
    }
  }
}

// -- ConvTranspose2d k4 s2 p1: 4 parity classes, tap-major implicit GEMM ------
// iy = ty + (jj>>1) - 1 + sy, ix = tx + (jj&1) - 1 + sx; out at (2ty+sy,2tx+sx).
__global__ void k_convT(const __bf16* __restrict__ Wp, const float* __restrict__ bias,
                        const __bf16* __restrict__ y2, float* __restrict__ out) {
  const int lane = threadIdx.x & 31;
  const int tile = blockIdx.x * 4 + (threadIdx.x >> 5);
  const int MT = OC / 16;                 // 6
  const int TOT = NB * 4 * MT * NT64;
  if (tile >= TOT) return;
  int t = tile;
  const int b = t / (4 * MT * NT64); t -= b * (4 * MT * NT64);
  const int s = t / (MT * NT64);     t -= s * (MT * NT64);
  const int m0 = (t / NT64) * 16;
  const int n0 = (t % NT64) * 64;
  const int sy = s >> 1, sx = s & 1;
  const int kh = (lane & 16);
  int ty[4], tx[4];
#pragma unroll
  for (int nn = 0; nn < 4; ++nn) {
    const int col = n0 + 16 * nn + (lane & 15);
    ty[nn] = col / WWI;
    tx[nn] = col % WWI;
  }
  const __bf16* A = Wp + (size_t)s * OC * (4 * DM);
  const __bf16* ib = y2 + (size_t)b * LL * DM;
  const __bf16* pa = A + (size_t)(m0 + (lane & 15)) * (4 * DM);
  f32x8 acc[4] = {};
#pragma unroll 1
  for (int jj = 0; jj < 4; ++jj) {
    if (jj < 3) __builtin_prefetch(pa + (jj + 1) * DM, 0, 0);
    const int dy = (jj >> 1) - 1 + sy, dx = (jj & 1) - 1 + sx;
    const __bf16* pb[4];
    bool ok[4];
#pragma unroll
    for (int nn = 0; nn < 4; ++nn) {
      const int iy = ty[nn] + dy, ix = tx[nn] + dx;
      ok[nn] = ((unsigned)iy < (unsigned)HH) && ((unsigned)ix < (unsigned)WWI);
      pb[nn] = ib + ((size_t)(iy * WWI + ix)) * DM + kh;
    }
#pragma unroll
    for (int c0 = 0; c0 < DM; c0 += 32) {
      const bf16x16 af = load_a_bf16(A, 4 * DM, m0, jj * DM + c0, lane);
#pragma unroll
      for (int nn = 0; nn < 4; ++nn) {
        bf16x16 bv = {};
        if (ok[nn])
          bv = concat8(*(const bf16x8*)(pb[nn] + c0), *(const bf16x8*)(pb[nn] + c0 + 8));
        WMMA_BF16(acc[nn], af, bv);
      }
    }
  }
  const int rb = m0 + ((lane & 16) ? 8 : 0);
#pragma unroll
  for (int nn = 0; nn < 4; ++nn) {
#pragma unroll
    for (int rr = 0; rr < 8; ++rr) {
      const int co = rb + rr;
      out[(((size_t)b * OC + co) * OH + (2 * ty[nn] + sy)) * OWD + (2 * tx[nn] + sx)] =
          acc[nn][rr] + bias[co];
    }
  }
}

}  // namespace

extern "C" void kernel_launch(void* const* d_in, const int* in_sizes, int n_in,
                              void* d_out, int out_size, void* d_ws, size_t ws_size,
                              hipStream_t stream) {
  (void)in_sizes; (void)n_in; (void)out_size; (void)ws_size;
  const float* x       = (const float*)d_in[0];
  const float* x_last  = (const float*)d_in[1];
  const float* in_w    = (const float*)d_in[2];
  const float* c1xw    = (const float*)d_in[3];
  const float* c1xb    = (const float*)d_in[4];
  const float* c1zw    = (const float*)d_in[5];
  const float* c1zb    = (const float*)d_in[6];
  const float* xproj_w = (const float*)d_in[7];
  const float* dt_w    = (const float*)d_in[8];
  const float* dt_b    = (const float*)d_in[9];
  const float* A_log   = (const float*)d_in[10];
  const float* ssm_D   = (const float*)d_in[11];
  const float* out_w   = (const float*)d_in[12];
  const float* c2w     = (const float*)d_in[13];
  const float* c2b     = (const float*)d_in[14];
  const float* ctw     = (const float*)d_in[15];
  const float* ctb     = (const float*)d_in[16];

  // ---------------- workspace layout (all activations channel-last) ----------
  const size_t E_XZ = (size_t)NB * LL * DM;   // 4,816,896
  const size_t E_HC = (size_t)NB * LL * DX;   // 2,408,448
  const size_t E_DB = (size_t)NB * LL * JP;   // 1,204,224
  __bf16* bw = (__bf16*)d_ws;
  __bf16* xbf   = bw;              bw += E_XZ;          // later reused as img_bf
  __bf16* xh_bf = bw;              bw += E_HC;
  __bf16* yz_bf = bw;              bw += E_XZ;          // later reused as y2_bf
  __bf16* w_in  = bw;              bw += DM * DM;       // 36,864
  __bf16* w_xp  = bw;              bw += JP * DX;       // 4,608
  __bf16* w_out = bw;              bw += DM * DM;       // 36,864
  __bf16* w_c2  = bw;              bw += DM * 9 * DM;   // 331,776
  __bf16* w_ct  = bw;              bw += 4 * OC * 4 * DM; // 294,912
  float* fw = (float*)bw;
  float* xz   = fw;                fw += E_XZ;
  float* xh_f = fw;                fw += E_HC;
  float* dtc  = fw;                fw += E_HC;
  float* xdbl = fw;                fw += E_DB;
  __bf16* img_bf = xbf;            // x_bf dead after k_inproj
  __bf16* y2_bf  = yz_bf;          // yz dead after k_outproj

  // ---------------- pre-pass: convert/pack to bf16 ----------------
  k_cvtT_x<<<(int)((E_XZ + 255) / 256), 256, 0, stream>>>(x, xbf);
  k_cvt<<<(DM * DM + 255) / 256, 256, 0, stream>>>(in_w, w_in, DM * DM);
  k_cvt<<<(DM * DM + 255) / 256, 256, 0, stream>>>(out_w, w_out, DM * DM);
  k_pack_xproj<<<(JP * DX + 255) / 256, 256, 0, stream>>>(xproj_w, w_xp);
  k_pack_c2<<<(DM * 9 * DM + 255) / 256, 256, 0, stream>>>(c2w, w_c2);
  k_pack_convT<<<(4 * OC * 4 * DM + 255) / 256, 256, 0, stream>>>(ctw, w_ct);

  // ---------------- main pipeline ----------------
  const int gemm_tiles = NB * (DM / 16) * NT64;        // 4704
  k_inproj <<<(gemm_tiles + 3) / 4, 128, 0, stream>>>(xbf, w_in, xz);
  k_dwconv <<<(int)((E_XZ + 255) / 256), 256, 0, stream>>>(
      xz, c1xw, c1xb, c1zw, c1zb, xh_f, xh_bf, yz_bf);
  k_xproj  <<<(NB * (JP / 16) * NT64 + 3) / 4, 128, 0, stream>>>(w_xp, xh_bf, xdbl);
  k_dt     <<<(int)((E_HC + 255) / 256), 256, 0, stream>>>(xdbl, dt_w, dt_b, dtc);
  k_scan   <<<NB, DX, 0, stream>>>(dtc, xh_f, xdbl, A_log, ssm_D, yz_bf);
  k_outproj<<<(gemm_tiles + 3) / 4, 128, 0, stream>>>(w_out, yz_bf, x_last, img_bf);
  k_conv2d <<<(gemm_tiles + 3) / 4, 128, 0, stream>>>(w_c2, c2b, img_bf, y2_bf);
  k_convT  <<<(NB * 4 * (OC / 16) * NT64 + 3) / 4, 128, 0, stream>>>(
      w_ct, ctb, y2_bf, (float*)d_out);
}
